// SwitchTransformersSparseMLP_10831907520601
// MI455X (gfx1250) — compile-verified, matrix operand order checked by reading, not run on hardware
//
#include <hip/hip_runtime.h>

// ---------------------------------------------------------------------------
// Switch-Transformers top-1 MoE for MI455X (gfx1250, wave32, WMMA).
//
// Roofline: 19.3 GFLOP useful work vs ~170 MB mandatory HBM traffic (weights
// are fp32 in memory) => memory floor ~7 us @ 23.3 TB/s.  bf16 WMMA
// (v_wmma_f32_16x16x32_bf16, f32 accum) keeps compute (~4 us) under that
// floor; f32 WMMA (K=4) would be ~8x slower and compute-bound.  All 151 MB
// of weights fit in the 192 MB L2, so per-tile weight re-reads hit L2.
//
// Round-2 changes: branchless, vectorized LDS staging (float4/uint2 global
// loads, packed b32/b64 LDS stores, token reads hoisted out of the K loop)
// so instruction issue doesn't throttle the memory pipes.
// ---------------------------------------------------------------------------

typedef __attribute__((ext_vector_type(16))) __bf16 v16bf;
typedef __attribute__((ext_vector_type(8)))  float  v8f;

#define D_MODEL 768
#define D_FF    3072
#define NE      8
#define T_TOK   2048          // 2 * 1024 tokens
#define TM      32            // tokens per workgroup tile
#define KB      32            // K step = WMMA K for bf16
#define KPAD    36            // LDS row stride 72B: 8B-aligned, conflict-free
#define MT_MAX  (T_TOK / TM)  // worst-case m-tiles per expert

union FragBF { unsigned u[8]; v16bf v; };

__device__ __forceinline__ unsigned short f2bf(float f) {
  // round-to-nearest-even fp32 -> bf16
  unsigned u = __float_as_uint(f);
  return (unsigned short)((u + 0x7FFFu + ((u >> 16) & 1u)) >> 16);
}
__device__ __forceinline__ unsigned pack2bf(float lo, float hi) {
  return (unsigned)f2bf(lo) | ((unsigned)f2bf(hi) << 16);
}

// ---------------------------------------------------------------------------
// Kernel 0: zero the per-expert token counters (workspace state must be
// reset every call; the harness does not re-poison between replays).
// ---------------------------------------------------------------------------
__global__ void zero_counts(int* __restrict__ counts) {
  if (threadIdx.x < NE) counts[threadIdx.x] = 0;
}

// ---------------------------------------------------------------------------
// Kernel 1: router.  logits = x @ Rw (fp32), p = max softmax prob = 1/sum,
// e = first-argmax (matches jnp.argmax tie-break).  Buckets token ids per
// expert via atomics; bucket order is nondeterministic but every token's
// downstream math is independent of its slot, so the output is deterministic.
// ---------------------------------------------------------------------------
__global__ __launch_bounds__(256)
void router_kernel(const float* __restrict__ x, const float* __restrict__ rw,
                   float* __restrict__ pvals, int* __restrict__ counts,
                   int* __restrict__ bucket) {
  const int t = blockIdx.x * blockDim.x + threadIdx.x;
  if (t >= T_TOK) return;
  float acc[NE];
#pragma unroll
  for (int e = 0; e < NE; ++e) acc[e] = 0.f;
  const float* xr = x + (size_t)t * D_MODEL;
  for (int k = 0; k < D_MODEL; ++k) {
    const float xv = xr[k];
    const float* r = rw + (size_t)k * NE;   // broadcast across threads, L2-hot
#pragma unroll
    for (int e = 0; e < NE; ++e) acc[e] = fmaf(xv, r[e], acc[e]);
  }
  float lmax = acc[0];
  int   ei   = 0;
#pragma unroll
  for (int e = 1; e < NE; ++e)
    if (acc[e] > lmax) { lmax = acc[e]; ei = e; }
  float s = 0.f;
#pragma unroll
  for (int e = 0; e < NE; ++e) s += __expf(acc[e] - lmax);
  pvals[t] = 1.0f / s;                        // max softmax probability
  const int pos = atomicAdd(&counts[ei], 1);
  bucket[ei * T_TOK + pos] = t;
}

// ---------------------------------------------------------------------------
// Grouped GEMM over routed tokens.
//   FIRST=true : D(bf16) = relu( gather_bf16(x) @ bf16(Wi_e) )  K=768,  N=3072
//   FIRST=false: out     = p .* ( H(bf16)     @ bf16(Wo_e) )    K=3072, N=768
// 256 threads = 8 waves.  M=32 (2 wmma m-tiles): wave (w&1) owns an m-half,
// wave (w>>1) owns an n-strip of TN/4 columns -> TN/64 wmma tiles per wave.
// A and B staged in LDS as bf16; B transposed to [n][k] so fragment loads are
// paired ds_load_b32 matching the CDNA5 16-bit operand layout.
// Staging assignment (hoisted out of the K loop):
//   A: thread (tid>>3) owns row, (tid&7)*4 owns 4 k's  -> 1 float4/uint2 load
//   B: thread owns LDS column n = tid%TN, KB*TN/256 k's -> coalesced b32
//      loads across lanes, packed-pair b32 stores along a contiguous LDS row
// ---------------------------------------------------------------------------
template <int TN, int K, int LDA, int LDB, bool FIRST>
__global__ __launch_bounds__(256)
void moe_gemm_kernel(const void*  __restrict__ Ag,      // x (f32) or H (bf16)
                     const float* __restrict__ Bg,      // wi or wo, fp32
                     const int*   __restrict__ counts,
                     const int*   __restrict__ bucket,
                     const float* __restrict__ pvals,
                     void*        __restrict__ Dg) {    // H (bf16) or out (f32)
  constexpr int NT_WAVE = TN / 16 / 4;                  // wmma n-tiles per wave
  constexpr int BKN     = KB * TN / 256;                // B k's per thread

  const int e   = blockIdx.x / MT_MAX;
  const int mt  = blockIdx.x % MT_MAX;
  const int n0  = blockIdx.y * TN;
  const int cnt = counts[e];
  if (mt * TM >= cnt) return;                           // empty tile -> exit

  __shared__ alignas(16) int            toks[TM];
  __shared__ alignas(16) float          pv[TM];
  __shared__ alignas(16) unsigned short Asm_[TM][KPAD];
  __shared__ alignas(16) unsigned short Bsm_[TN][KPAD];

  const int tid = threadIdx.x;
  if (tid < TM) {
    const int row = mt * TM + tid;
    const int tok = (row < cnt) ? bucket[e * T_TOK + row] : -1;
    toks[tid] = tok;
    pv[tid]   = (!FIRST && tok >= 0) ? pvals[tok] : 0.f;
  }
  __syncthreads();

  // ---- hoisted staging coordinates ----
  const int ar = tid >> 3;               // A row this thread stages
  const int ak = (tid & 7) << 2;         // 4 consecutive k's
  const int tokA = toks[ar];
  const size_t tokSafe = (size_t)(tokA < 0 ? 0 : tokA);
  const float*          Arow_f = (const float*)Ag + tokSafe * LDA;
  const unsigned short* Arow_h = (const unsigned short*)Ag + tokSafe * LDA;

  const int bn  = tid % TN;              // B column this thread stages
  const int bkb = (tid / TN) * BKN;      // first k of this thread's segment
  const float* Bcol = Bg + (size_t)e * K * LDB + n0 + bn;

  const int w     = tid >> 5;
  const int lane  = tid & 31;
  const int mloc  = (w & 1) * 16;        // which 16-row half of the M tile
  const int wc    = w >> 1;              // wave column 0..3
  const int l15   = lane & 15;
  const int khalf = (lane >> 4) * 8;     // lanes 16-31 hold K+8 (ISA layout)

  v8f acc[NT_WAVE];
#pragma unroll
  for (int j = 0; j < NT_WAVE; ++j)
#pragma unroll
    for (int i = 0; i < 8; ++i) acc[j][i] = 0.f;

  for (int k0 = 0; k0 < K; k0 += KB) {
    // ---- stage A: one vector load + two packed b32 stores, branchless ----
    {
      unsigned p0, p1;
      if constexpr (FIRST) {
        float4 v = *(const float4*)&Arow_f[k0 + ak];     // 16B aligned
        if (tokA < 0) { v.x = 0.f; v.y = 0.f; v.z = 0.f; v.w = 0.f; }
        p0 = pack2bf(v.x, v.y);
        p1 = pack2bf(v.z, v.w);
      } else {
        uint2 v = *(const uint2*)&Arow_h[k0 + ak];       // 8B aligned bf16x4
        if (tokA < 0) { v.x = 0u; v.y = 0u; }
        p0 = v.x;
        p1 = v.y;
      }
      *(unsigned*)&Asm_[ar][ak]     = p0;
      *(unsigned*)&Asm_[ar][ak + 2] = p1;
    }
    // ---- stage B: coalesced fp32 column loads, packed-pair b32 stores ----
#pragma unroll
    for (int i = 0; i < BKN; i += 2) {
      const int kk = bkb + i;
      const float lo = Bcol[(size_t)(k0 + kk) * LDB];
      const float hi = Bcol[(size_t)(k0 + kk + 1) * LDB];
      *(unsigned*)&Bsm_[bn][kk] = pack2bf(lo, hi);
    }
    // prefetch next weight K-slab into L2 (emits global_prefetch_b8)
    if (k0 + KB < K)
      __builtin_prefetch(&Bcol[(size_t)(k0 + KB + (tid & 31)) * LDB], 0, 1);
    __syncthreads();

    // ---- fragments + WMMA ----
    FragBF a;
#pragma unroll
    for (int r = 0; r < 8; ++r) {
      const int k = khalf + ((r & 3) << 1) + ((r >> 2) << 4); // K pair index
      a.u[r] = *(const unsigned*)&Asm_[mloc + l15][k];
    }
#pragma unroll
    for (int j = 0; j < NT_WAVE; ++j) {
      const int nb = (wc * NT_WAVE + j) * 16;
      FragBF b;
#pragma unroll
      for (int r = 0; r < 8; ++r) {
        const int k = khalf + ((r & 3) << 1) + ((r >> 2) << 4);
        b.u[r] = *(const unsigned*)&Bsm_[nb + l15][k];
      }
      acc[j] = __builtin_amdgcn_wmma_f32_16x16x32_bf16(
          /*neg_a=*/false, a.v, /*neg_b=*/false, b.v,
          /*c_mod=*/(short)0, acc[j], /*reuse_a=*/false, /*reuse_b=*/false);
    }
    __syncthreads();
  }

  // ---- epilogue: C layout = VGPR cr -> M=cr (lanes 0-15) / M=8+cr (16-31) ----
#pragma unroll
  for (int j = 0; j < NT_WAVE; ++j) {
    const int ncol = n0 + (wc * NT_WAVE + j) * 16 + l15;
#pragma unroll
    for (int cr = 0; cr < 8; ++cr) {
      const int row = mloc + cr + ((lane >> 4) << 3);
      const int tok = toks[row];
      if (tok < 0) continue;
      const float val = acc[j][cr];
      if constexpr (FIRST) {
        ((unsigned short*)Dg)[(size_t)tok * D_FF + ncol] =
            f2bf(val > 0.f ? val : 0.f);                 // relu -> bf16 H
      } else {
        ((float*)Dg)[(size_t)tok * D_MODEL + ncol] = pv[row] * val;
      }
    }
  }
}

// ---------------------------------------------------------------------------
// Workspace layout (needs ~12.7 MB):
//   [0      ..   32) per-expert counts (8 x i32)
//   [1024   .. 9216) pvals   (2048 x f32)
//   [16384  .. 81920) bucket (8 x 2048 x i32)
//   [131072 .. +12.6MB) H    (2048 x 3072 bf16)
// ---------------------------------------------------------------------------
extern "C" void kernel_launch(void* const* d_in, const int* in_sizes, int n_in,
                              void* d_out, int out_size, void* d_ws, size_t ws_size,
                              hipStream_t stream) {
  const float* x  = (const float*)d_in[0];   // [2,1024,768]
  const float* rw = (const float*)d_in[1];   // [768,8]
  const float* wi = (const float*)d_in[2];   // [8,768,3072]
  const float* wo = (const float*)d_in[3];   // [8,3072,768]
  float* out = (float*)d_out;

  char* ws = (char*)d_ws;
  int*            counts = (int*)(ws + 0);
  float*          pvals  = (float*)(ws + 1024);
  int*            bucket = (int*)(ws + 16384);
  unsigned short* H      = (unsigned short*)(ws + 131072);

  zero_counts<<<1, 32, 0, stream>>>(counts);
  router_kernel<<<T_TOK / 256, 256, 0, stream>>>(x, rw, pvals, counts, bucket);

  // GEMM1: H = relu(Xg @ Wi_e)   M<=2048, N=3072, K=768
  moe_gemm_kernel<256, D_MODEL, D_MODEL, D_FF, true>
      <<<dim3(NE * MT_MAX, D_FF / 256), 256, 0, stream>>>(
          x, wi, counts, bucket, nullptr, H);

  // GEMM2: out = p .* (Hg @ Wo_e)  M<=2048, N=768, K=3072
  moe_gemm_kernel<128, D_FF, D_FF, D_MODEL, false>
      <<<dim3(NE * MT_MAX, D_MODEL / 128), 256, 0, stream>>>(
          H, wo, counts, bucket, pvals, out);
}
